// AttentiveFPModel_11733850653138
// MI455X (gfx1250) — compile-verified, matrix operand order checked by reading, not run on hardware
//
#include <hip/hip_runtime.h>
#include <hip/hip_bf16.h>

#define H 128
#define NEG 0.01f

typedef __attribute__((ext_vector_type(16))) __bf16 v16bf;
typedef __attribute__((ext_vector_type(8)))  float  v8f;

// ---- order-preserving float<->uint key for atomic float max via atomicMax(u32) ----
__device__ __forceinline__ unsigned fkey(float f) {
    unsigned u = __float_as_uint(f);
    return (u >> 31) ? ~u : (u | 0x80000000u);
}
__device__ __forceinline__ float funkey(unsigned k) {
    return (k >> 31) ? __uint_as_float(k ^ 0x80000000u) : __uint_as_float(~k);
}
__device__ __forceinline__ float lrelu(float v) { return v > 0.f ? v : NEG * v; }

// =====================================================================
// WMMA GEMM:  C[n, m] = sum_k X[n, k] * W[m, k]      (X:[nrows,128] f32,
// W:[M, ldw] f32 row-major, C:[nrows, M] f32).  bf16 inputs, f32 accum.
// One wave -> one 16x16 tile. blockDim=256 (8 waves); grid=(nrows/16, M/128).
// A layout (16-bit 16x32): lane m=L&15, khalf=L>>4; elems 0..7 <-> k=khalf*8+i,
// elems 8..15 <-> k=16+khalf*8+i.  B mirrored with lane n = L&15 = W row.
// C/D: lane L: col n=L&15, VGPR r -> row m = 8*(L>>4)+r.
// =====================================================================
__global__ void k_gemm(const float* __restrict__ Xm, const float* __restrict__ Wm,
                       float* __restrict__ Cm, int ldw, int M, int nrows) {
    int wave = threadIdx.x >> 5;
    int lane = threadIdx.x & 31;
    int row0 = blockIdx.x * 16;
    int f0   = blockIdx.y * 128 + wave * 16;
    if (row0 >= nrows || f0 >= M) return;
    int mrow  = lane & 15;
    int khalf = lane >> 4;
    const float* xp = Xm + (size_t)(row0 + mrow) * H  + khalf * 8;
    const float* wp = Wm + (size_t)(f0   + mrow) * ldw + khalf * 8;
    v8f acc = {};
#pragma unroll
    for (int kb = 0; kb < 128; kb += 32) {
        v16bf a, b;
#pragma unroll
        for (int i = 0; i < 8; ++i) {
            a[i]     = (__bf16)xp[kb + i];
            a[8 + i] = (__bf16)xp[kb + 16 + i];
            b[i]     = (__bf16)wp[kb + i];
            b[8 + i] = (__bf16)wp[kb + 16 + i];
        }
        acc = __builtin_amdgcn_wmma_f32_16x16x32_bf16(false, a, false, b,
                                                      (short)0, acc, false, false);
    }
    int ncol  = lane & 15;
    int mbase = khalf * 8;
#pragma unroll
    for (int r = 0; r < 8; ++r)
        Cm[(size_t)(row0 + mbase + r) * M + f0 + ncol] = acc[r];
}

// ---- lin1: x = lrelu(node_attr * W_lin1 + b) ------------------------
__global__ void k_lin1(const float* __restrict__ na, const float* __restrict__ w,
                       const float* __restrict__ b, float* __restrict__ x, int n) {
    int t = blockIdx.x * blockDim.x + threadIdx.x;
    if (t >= n * H) return;
    int i = t >> 7, k = t & 127;
    x[t] = lrelu(na[i] * w[k] + b[k]);
}

// ---- per-node dot(s) with attention vectors (wave per node) ---------
__global__ void k_dot1(const float* __restrict__ Xm, const float* __restrict__ a1,
                       float* __restrict__ o1, int n) {
    int wid = (blockIdx.x * blockDim.x + threadIdx.x) >> 5;
    int lane = threadIdx.x & 31;
    if (wid >= n) return;
    float4 xv = ((const float4*)(Xm + (size_t)wid * H))[lane];
    float4 v1 = ((const float4*)a1)[lane];
    float s = xv.x * v1.x + xv.y * v1.y + xv.z * v1.z + xv.w * v1.w;
    for (int off = 16; off; off >>= 1) s += __shfl_xor(s, off, 32);
    if (lane == 0) o1[wid] = s;
}
__global__ void k_dot2(const float* __restrict__ Xm, const float* __restrict__ a1,
                       const float* __restrict__ a2, float* __restrict__ o1,
                       float* __restrict__ o2, int n) {
    int wid = (blockIdx.x * blockDim.x + threadIdx.x) >> 5;
    int lane = threadIdx.x & 31;
    if (wid >= n) return;
    float4 xv = ((const float4*)(Xm + (size_t)wid * H))[lane];
    float4 v1 = ((const float4*)a1)[lane];
    float4 v2 = ((const float4*)a2)[lane];
    float s1 = xv.x * v1.x + xv.y * v1.y + xv.z * v1.z + xv.w * v1.w;
    float s2 = xv.x * v2.x + xv.y * v2.y + xv.z * v2.z + xv.w * v2.w;
    for (int off = 16; off; off >>= 1) { s1 += __shfl_xor(s1, off, 32); s2 += __shfl_xor(s2, off, 32); }
    if (lane == 0) { o1[wid] = s1; o2[wid] = s2; }
}

// ---- GATEConv edge logits (wave per edge) ---------------------------
// logit = lrelu( dot(lrelu(P[src]+ea*w_e), att_l) + r_dst[dst] ); atomic max key
__global__ void k_edge_gate_logit(const int* __restrict__ ei, const float* __restrict__ ea,
                                  const float* __restrict__ P, const float* __restrict__ gW1,
                                  const float* __restrict__ att_l, const float* __restrict__ rdst,
                                  float* __restrict__ logit, unsigned* __restrict__ mkey, int E_) {
    int wid = (blockIdx.x * blockDim.x + threadIdx.x) >> 5;
    int lane = threadIdx.x & 31;
    if (wid >= E_) return;
    int s = ei[wid], d = ei[E_ + wid];
    float eaw = ea[wid];
    float4 pv = ((const float4*)(P + (size_t)s * H))[lane];
    float4 al = ((const float4*)att_l)[lane];
    int k0 = lane * 4;
    float w0 = gW1[(size_t)(k0 + 0) * 129 + 128];
    float w1 = gW1[(size_t)(k0 + 1) * 129 + 128];
    float w2 = gW1[(size_t)(k0 + 2) * 129 + 128];
    float w3 = gW1[(size_t)(k0 + 3) * 129 + 128];
    float sum = lrelu(pv.x + eaw * w0) * al.x + lrelu(pv.y + eaw * w1) * al.y +
                lrelu(pv.z + eaw * w2) * al.z + lrelu(pv.w + eaw * w3) * al.w;
    for (int off = 16; off; off >>= 1) sum += __shfl_xor(sum, off, 32);
    if (lane == 0) {
        float r = lrelu(sum + rdst[d]);
        logit[wid] = r;
        atomicMax(&mkey[d], fkey(r));
    }
}

// ---- simple edge logits (GATConv): one thread per edge --------------
__global__ void k_edge_slogit(const int* __restrict__ ei, const float* __restrict__ ss,
                              const float* __restrict__ sd, float* __restrict__ logit,
                              unsigned* __restrict__ mkey, int E_) {
    int e = blockIdx.x * blockDim.x + threadIdx.x;
    if (e >= E_) return;
    int d = ei[E_ + e];
    float v = lrelu(ss[ei[e]] + sd[d]);
    logit[e] = v;
    atomicMax(&mkey[d], fkey(v));
}

// ---- softmax-weighted aggregation (wave per edge) -------------------
// e = exp(logit - max[dst]); s[dst]+=e; hacc[dst][k] += e*F[src][k]
__global__ void k_edge_accum(const int* __restrict__ ei, const float* __restrict__ logit,
                             const unsigned* __restrict__ mkey, float* __restrict__ ssum,
                             const float* __restrict__ F, float* __restrict__ hacc, int E_) {
    int wid = (blockIdx.x * blockDim.x + threadIdx.x) >> 5;
    int lane = threadIdx.x & 31;
    if (wid >= E_) return;
    int s = ei[wid], d = ei[E_ + wid];
    float ex = __expf(logit[wid] - funkey(mkey[d]));
    if (lane == 0) atomicAdd(&ssum[d], ex);
    float4 fv = ((const float4*)(F + (size_t)s * H))[lane];
    float* hp = hacc + (size_t)d * H + lane * 4;
    atomicAdd(hp + 0, ex * fv.x);
    atomicAdd(hp + 1, ex * fv.y);
    atomicAdd(hp + 2, ex * fv.z);
    atomicAdd(hp + 3, ex * fv.w);
}

// ---- h = elu(hacc/s + bias)  (guard empty segments) -----------------
__global__ void k_node_h(float* __restrict__ hacc, const float* __restrict__ ssum,
                         const float* __restrict__ bias, int n) {
    int t = blockIdx.x * blockDim.x + threadIdx.x;
    if (t >= n * H) return;
    int i = t >> 7, k = t & 127;
    float s = ssum[i];
    float v = (s > 0.f ? hacc[t] / s : 0.f) + bias[k];
    hacc[t] = v > 0.f ? v : (__expf(v) - 1.f);
}

// ---- GRU gates + ReLU, in-place on hidden x -------------------------
__global__ void k_gru_relu(const float* __restrict__ GI, const float* __restrict__ GH,
                           const float* __restrict__ bih, const float* __restrict__ bhh,
                           float* __restrict__ x, int n) {
    int t = blockIdx.x * blockDim.x + threadIdx.x;
    if (t >= n * H) return;
    int i = t >> 7, k = t & 127;
    size_t b = (size_t)i * 384;
    float ir = GI[b + k] + bih[k],       hr = GH[b + k] + bhh[k];
    float iz = GI[b + 128 + k] + bih[128 + k], hz = GH[b + 128 + k] + bhh[128 + k];
    float in = GI[b + 256 + k] + bih[256 + k], hn = GH[b + 256 + k] + bhh[256 + k];
    float r = 1.f / (1.f + __expf(-(ir + hr)));
    float z = 1.f / (1.f + __expf(-(iz + hz)));
    float nn = tanhf(in + r * hn);
    float o = (1.f - z) * nn + z * x[t];
    x[t] = o > 0.f ? o : 0.f;
}

// ---- mol pooling helpers --------------------------------------------
// mol layout (floats): [0..127]=colsum, [128..255]=out0, [256..383]=outs,
//                      [384..511]=hacc_m, [512]=s_mol, [513]=max key(u32), [514]=c
__global__ void k_colsum(const float* __restrict__ x, float* __restrict__ mol, int n) {
    int k = threadIdx.x;
    float acc = 0.f;
    for (int i = blockIdx.x; i < n; i += gridDim.x) acc += x[(size_t)i * H + k];
    atomicAdd(&mol[k], acc);
}
__global__ void k_outs_c(float* __restrict__ mol, const float* __restrict__ molW,
                         const float* __restrict__ attd) {
    __shared__ float red[H];
    int t = threadIdx.x;
    float o = mol[t];
    mol[128 + t] = o > 0.f ? o : 0.f;     // out0 = relu(sum)
    __syncthreads();
    float s = 0.f;
    for (int j = 0; j < H; ++j) s += mol[128 + j] * molW[(size_t)t * H + j];
    mol[256 + t] = s;                      // outs
    red[t] = s * attd[t];
    __syncthreads();
    for (int off = 64; off; off >>= 1) { if (t < off) red[t] += red[t + off]; __syncthreads(); }
    if (t == 0) mol[514] = red[0];         // c = dot(outs, att_dst)
}
__global__ void k_mol_logit(const float* __restrict__ xs, const float* __restrict__ att,
                            float* __restrict__ mol, float* __restrict__ logit, int n) {
    int wid = (blockIdx.x * blockDim.x + threadIdx.x) >> 5;
    int lane = threadIdx.x & 31;
    if (wid >= n) return;
    float4 xv = ((const float4*)(xs + (size_t)wid * H))[lane];
    float4 av = ((const float4*)att)[lane];
    float s = xv.x * av.x + xv.y * av.y + xv.z * av.z + xv.w * av.w;
    for (int off = 16; off; off >>= 1) s += __shfl_xor(s, off, 32);
    if (lane == 0) {
        float v = lrelu(s + mol[514]);
        logit[wid] = v;
        atomicMax(((unsigned*)mol) + 513, fkey(v));
    }
}
__global__ void k_mol_accum(const float* __restrict__ xs, const float* __restrict__ logit,
                            float* __restrict__ mol, int n) {
    int k = threadIdx.x;
    float m = funkey(((const unsigned*)mol)[513]);
    float acc = 0.f, esum = 0.f;
    for (int i = blockIdx.x; i < n; i += gridDim.x) {
        float e = __expf(logit[i] - m);
        acc += e * xs[(size_t)i * H + k];
        esum += e;
    }
    atomicAdd(&mol[384 + k], acc);
    if (k == 0) atomicAdd(&mol[512], esum);
}
__global__ void k_mol_final(const float* __restrict__ mol, const float* __restrict__ molb,
                            const float* __restrict__ Wih, const float* __restrict__ Whh,
                            const float* __restrict__ bih, const float* __restrict__ bhh,
                            float* __restrict__ out) {
    __shared__ float hm[H], h0[H];
    int t = threadIdx.x;
    float s = mol[512];
    float v = (s > 0.f ? mol[384 + t] / s : 0.f) + molb[t];
    hm[t] = v > 0.f ? v : (__expf(v) - 1.f);
    h0[t] = mol[128 + t];
    __syncthreads();
    float ir = bih[t], iz = bih[128 + t], in = bih[256 + t];
    float hr = bhh[t], hz = bhh[128 + t], hn = bhh[256 + t];
    for (int j = 0; j < H; ++j) {
        float hj = hm[j], oj = h0[j];
        ir += hj * Wih[(size_t)t * H + j];
        iz += hj * Wih[(size_t)(128 + t) * H + j];
        in += hj * Wih[(size_t)(256 + t) * H + j];
        hr += oj * Whh[(size_t)t * H + j];
        hz += oj * Whh[(size_t)(128 + t) * H + j];
        hn += oj * Whh[(size_t)(256 + t) * H + j];
    }
    float r = 1.f / (1.f + __expf(-(ir + hr)));
    float z = 1.f / (1.f + __expf(-(iz + hz)));
    float nn = tanhf(in + r * hn);
    float o = (1.f - z) * nn + z * h0[t];
    out[t] = o > 0.f ? o : 0.f;
}

extern "C" void kernel_launch(void* const* d_in, const int* in_sizes, int n_in,
                              void* d_out, int out_size, void* d_ws, size_t ws_size,
                              hipStream_t stream) {
    const float* node_attr = (const float*)d_in[0];
    const int*   ei        = (const int*)d_in[1];
    const float* edge_attr = (const float*)d_in[2];
    const float* W_lin1    = (const float*)d_in[3];
    const float* b_lin1    = (const float*)d_in[4];
    const float* gate_W1   = (const float*)d_in[5];   // [H, H+1]
    const float* gate_W2   = (const float*)d_in[6];
    const float* gate_attl = (const float*)d_in[7];
    const float* gate_attr = (const float*)d_in[8];
    const float* gate_b    = (const float*)d_in[9];
    const float* g1_Wih    = (const float*)d_in[10];
    const float* g1_Whh    = (const float*)d_in[11];
    const float* g1_bih    = (const float*)d_in[12];
    const float* g1_bhh    = (const float*)d_in[13];
    const float* atom_W    = (const float*)d_in[14];
    const float* atom_as   = (const float*)d_in[15];
    const float* atom_ad   = (const float*)d_in[16];
    const float* atom_b    = (const float*)d_in[17];
    const float* g2_Wih    = (const float*)d_in[18];
    const float* g2_Whh    = (const float*)d_in[19];
    const float* g2_bih    = (const float*)d_in[20];
    const float* g2_bhh    = (const float*)d_in[21];
    const float* mol_W     = (const float*)d_in[22];
    const float* mol_as    = (const float*)d_in[23];
    const float* mol_ad    = (const float*)d_in[24];
    const float* mol_b     = (const float*)d_in[25];
    const float* gm_Wih    = (const float*)d_in[26];
    const float* gm_Whh    = (const float*)d_in[27];
    const float* gm_bih    = (const float*)d_in[28];
    const float* gm_bhh    = (const float*)d_in[29];

    const int N_ = in_sizes[0];       // node_attr [N,1]
    const int E_ = in_sizes[2];       // edge_attr [E,1]
    const size_t NH = (size_t)N_ * H;

    // workspace layout (floats); L2-resident: each N*H table is 25.6 MB
    float* ws  = (float*)d_ws;
    float* X   = ws;                  // x            [N,H]
    float* T0  = X  + NH;             // P / xt / xs  [N,H]
    float* T1  = T0 + NH;             // G            [N,H]
    float* HH  = T1 + NH;             // hacc -> h    [N,H]
    float* GI  = HH + NH;             // gru input-gates  [N,3H]
    float* GH  = GI + NH * 3;         // gru hidden-gates [N,3H]
    float* LOG = GH + NH * 3;         // edge logits  [E]
    unsigned* MK = (unsigned*)(LOG + E_); // per-node max keys [N]
    float* S_  = (float*)(MK + N_);   // per-node exp sums [N]
    float* SS  = S_ + N_;             // r_dst / s_src [N]
    float* SD  = SS + N_;             // s_dst [N]
    float* MOL = SD + N_;             // mol scratch [1024]

    dim3 b256(256), b128(128);
    int gElem  = (int)((NH + 255) / 256);
    int gWaveN = (N_ + 7) / 8;        // wave-per-node, 8 waves/block
    int gWaveE = (E_ + 7) / 8;        // wave-per-edge
    int gThrE  = (E_ + 255) / 256;
    dim3 gT16(N_ / 16, 1);            // N = 50000 = 3125*16
    dim3 gT48(N_ / 16, 3);            // M = 384 GRU gate GEMMs

    // init accumulators (graph-capture-safe)
    hipMemsetAsync(MK, 0, (size_t)N_ * 4, stream);   // key 0 == -inf
    hipMemsetAsync(S_, 0, (size_t)N_ * 4, stream);
    hipMemsetAsync(HH, 0, NH * 4, stream);
    hipMemsetAsync(MOL, 0, 1024 * 4, stream);

    // ---- lin1 ----
    k_lin1<<<gElem, b256, 0, stream>>>(node_attr, W_lin1, b_lin1, X, N_);

    // ---- GATEConv ----
    k_gemm<<<gT16, b256, 0, stream>>>(X, gate_W1, T0, 129, 128, N_);   // P
    k_dot1<<<gWaveN, b256, 0, stream>>>(X, gate_attr, SS, N_);         // r_dst
    k_gemm<<<gT16, b256, 0, stream>>>(X, gate_W2, T1, 128, 128, N_);   // G
    k_edge_gate_logit<<<gWaveE, b256, 0, stream>>>(ei, edge_attr, T0, gate_W1,
                                                   gate_attl, SS, LOG, MK, E_);
    k_edge_accum<<<gWaveE, b256, 0, stream>>>(ei, LOG, MK, S_, T1, HH, E_);
    k_node_h<<<gElem, b256, 0, stream>>>(HH, S_, gate_b, N_);
    k_gemm<<<gT48, b256, 0, stream>>>(HH, g1_Wih, GI, 128, 384, N_);
    k_gemm<<<gT48, b256, 0, stream>>>(X,  g1_Whh, GH, 128, 384, N_);
    k_gru_relu<<<gElem, b256, 0, stream>>>(GI, GH, g1_bih, g1_bhh, X, N_);

    // ---- GATConv ----
    hipMemsetAsync(MK, 0, (size_t)N_ * 4, stream);
    hipMemsetAsync(S_, 0, (size_t)N_ * 4, stream);
    hipMemsetAsync(HH, 0, NH * 4, stream);
    k_gemm<<<gT16, b256, 0, stream>>>(X, atom_W, T0, 128, 128, N_);    // xt
    k_dot2<<<gWaveN, b256, 0, stream>>>(T0, atom_as, atom_ad, SS, SD, N_);
    k_edge_slogit<<<gThrE, b256, 0, stream>>>(ei, SS, SD, LOG, MK, E_);
    k_edge_accum<<<gWaveE, b256, 0, stream>>>(ei, LOG, MK, S_, T0, HH, E_);
    k_node_h<<<gElem, b256, 0, stream>>>(HH, S_, atom_b, N_);
    k_gemm<<<gT48, b256, 0, stream>>>(HH, g2_Wih, GI, 128, 384, N_);
    k_gemm<<<gT48, b256, 0, stream>>>(X,  g2_Whh, GH, 128, 384, N_);
    k_gru_relu<<<gElem, b256, 0, stream>>>(GI, GH, g2_bih, g2_bhh, X, N_);

    // ---- molecule embedding (single graph) ----
    k_colsum<<<512, b128, 0, stream>>>(X, MOL, N_);
    k_gemm<<<gT16, b256, 0, stream>>>(X, mol_W, T0, 128, 128, N_);     // xs
    k_outs_c<<<1, b128, 0, stream>>>(MOL, mol_W, mol_ad);
    k_mol_logit<<<gWaveN, b256, 0, stream>>>(T0, mol_as, MOL, LOG, N_);
    k_mol_accum<<<512, b128, 0, stream>>>(T0, LOG, MOL, N_);
    k_mol_final<<<1, b128, 0, stream>>>(MOL, mol_b, gm_Wih, gm_Whh,
                                        gm_bih, gm_bhh, (float*)d_out);
}
